// EncoderLayer_13073880449047
// MI455X (gfx1250) — matrix operand order repaired and checked by reading, not compiled, not run on hardware
//
#include <hip/hip_runtime.h>
#include <hip/hip_bf16.h>
#include <math.h>

// ---------------------------------------------------------------------------
// Problem constants (reference): B=2, S=2048, H=1024, NH=16, DH=64, FF=4096
// ---------------------------------------------------------------------------
#define S_LEN  2048
#define HID    1024
#define NHEAD  16
#define DHEAD  64
#define FFDIM  4096
#define NTOK   4096   // B*S

typedef __bf16 bf16_t;
typedef __attribute__((ext_vector_type(16))) __bf16 v16bf;
typedef __attribute__((ext_vector_type(8)))  float  v8f;

union FragBF { v16bf v; float4 f[2]; };

__device__ __forceinline__ v8f wmma_bf16(v16bf a, v16bf b, v8f c) {
  // D = A(16x32 bf16) * B(32x16 bf16) + C(16x16 f32)
  return __builtin_amdgcn_wmma_f32_16x16x32_bf16(
      /*neg_a=*/false, a, /*neg_b=*/false, b,
      /*c_mod=*/(short)0, c, /*reuse_a=*/false, /*reuse_b=*/false);
}

// Async global -> LDS copy, 16 bytes per lane, tracked by ASYNCcnt.
// LDS address = low 32 bits of the flat pointer.
__device__ __forceinline__ void async_b128(void* lds_dst, const void* gsrc) {
  unsigned l = (unsigned)(uintptr_t)lds_dst;
  asm volatile("global_load_async_to_lds_b128 %0, %1, off"
               :: "v"(l), "v"(gsrc) : "memory");
}
__device__ __forceinline__ void wait_async0() {
  asm volatile("s_wait_asynccnt 0x0" ::: "memory");
}

// ---------------------------------------------------------------------------
// f32 -> bf16 elementwise convert (grid-stride)
// ---------------------------------------------------------------------------
__global__ __launch_bounds__(256) void cvt_f32_bf16(const float* __restrict__ in,
                                                    bf16_t* __restrict__ out, int n) {
  int i = blockIdx.x * blockDim.x + threadIdx.x;
  int stride = gridDim.x * blockDim.x;
  for (; i < n; i += stride) out[i] = (bf16_t)in[i];
}

// ---------------------------------------------------------------------------
// f32 [K,N] -> bf16 [N,K] tiled transpose (weights, once per call).
// 32x32 tile, 256 threads.
// ---------------------------------------------------------------------------
__global__ __launch_bounds__(256) void cvt_transpose_bf16(
    const float* __restrict__ in, bf16_t* __restrict__ out, int K, int N) {
  __shared__ float tile[32][33];
  const int k0 = blockIdx.y * 32, n0 = blockIdx.x * 32;
  const int tx = threadIdx.x & 31, ty = threadIdx.x >> 5;  // ty 0..7
#pragma unroll
  for (int i = 0; i < 4; ++i) {
    int k = ty + i * 8;
    tile[k][tx] = in[(size_t)(k0 + k) * N + n0 + tx];
  }
  __syncthreads();
#pragma unroll
  for (int i = 0; i < 4; ++i) {
    int n = ty + i * 8;
    out[(size_t)(n0 + n) * K + k0 + tx] = (bf16_t)tile[tx][n];
  }
}

// ---------------------------------------------------------------------------
// bf16 WMMA GEMM:  C[M,N] = A[M,K] @ W^T  with W given pre-transposed [N,K].
// Block: 256 threads (8 wave32), tile 128(M) x 64(N), K step 32.
// Double-buffered LDS filled by global_load_async_to_lds_b128.
// OUT_MODE: 0 = f32 [M,N]; 1 = bf16 [M,N]; 2 = bf16 [B,NH,S,DH] (Q/K);
//           3 = bf16 [B,NH,DH,S] (V, pre-transposed for attention).
// ---------------------------------------------------------------------------
template <int OUT_MODE, bool GELU_ACT>
__global__ __launch_bounds__(256) void gemm_bf16_wmma(
    const bf16_t* __restrict__ A,    // [M,K]
    const bf16_t* __restrict__ Bt,   // [N,K]
    const float* __restrict__ bias,
    float* __restrict__ Cf, bf16_t* __restrict__ Cb,
    int M, int N, int K) {
  __shared__ bf16_t As[2][128][40];  // 128x32 halves (+8 pad, 16B-aligned rows)
  __shared__ bf16_t Bs[2][64][40];   // 64x32 halves  (+8 pad)

  const int tid    = threadIdx.x;
  const int lane   = tid & 31;
  const int wave   = tid >> 5;
  const int laneM  = lane & 15;
  const int laneHi = lane >> 4;
  const int wm     = (wave & 3) * 32;
  const int wn     = (wave >> 2) * 32;
  const int m0     = blockIdx.y * 128;
  const int n0     = blockIdx.x * 64;

  // per-thread staging assignment (3 async b128 per thread per tile)
  const int ar = tid >> 1, ac = (tid & 1) * 16;  // A: row 0..127, half-chunk
  const int br = tid >> 2, bc = (tid & 3) * 8;   // B: row 0..63

  const bf16_t* Abase = &A[(size_t)(m0 + ar) * K + ac];
  const bf16_t* Bbase = &Bt[(size_t)(n0 + br) * K + bc];

  v8f acc[2][2] = {};

  // preload tile 0
  async_b128(&As[0][ar][ac],     Abase);
  async_b128(&As[0][ar][ac + 8], Abase + 8);
  async_b128(&Bs[0][br][bc],     Bbase);

  const int nIter = K >> 5;
  for (int it = 0; it < nIter; ++it) {
    const int cur = it & 1;
    wait_async0();        // this wave's portion of tile `it` is in LDS
    __syncthreads();      // everyone's portion present; prev buffer reads done
    if (it + 1 < nIter) {
      const int k1 = (it + 1) << 5;
      async_b128(&As[cur ^ 1][ar][ac],     Abase + k1);
      async_b128(&As[cur ^ 1][ar][ac + 8], Abase + k1 + 8);
      async_b128(&Bs[cur ^ 1][br][bc],     Bbase + k1);
    }

    FragBF a[2], b[2];
#pragma unroll
    for (int mi = 0; mi < 2; ++mi) {
      int row = wm + mi * 16 + laneM;
      a[mi].f[0] = *(const float4*)&As[cur][row][laneHi * 8];
      a[mi].f[1] = *(const float4*)&As[cur][row][16 + laneHi * 8];
    }
#pragma unroll
    for (int ni = 0; ni < 2; ++ni) {
      int col = wn + ni * 16 + laneM;
      b[ni].f[0] = *(const float4*)&Bs[cur][col][laneHi * 8];
      b[ni].f[1] = *(const float4*)&Bs[cur][col][16 + laneHi * 8];
    }
#pragma unroll
    for (int mi = 0; mi < 2; ++mi)
#pragma unroll
      for (int ni = 0; ni < 2; ++ni)
        acc[mi][ni] = wmma_bf16(a[mi].v, b[ni].v, acc[mi][ni]);
  }

  // ---- epilogue ----
#pragma unroll
  for (int mi = 0; mi < 2; ++mi) {
#pragma unroll
    for (int ni = 0; ni < 2; ++ni) {
      int col = n0 + wn + ni * 16 + laneM;
      float bv = bias[col];
#pragma unroll
      for (int r = 0; r < 8; ++r) {
        int row = m0 + wm + mi * 16 + laneHi * 8 + r;
        float vv = acc[mi][ni][r] + bv;
        if constexpr (GELU_ACT)
          vv = 0.5f * vv * (1.0f + erff(vv * 0.7071067811865475f));
        if constexpr (OUT_MODE == 0) {
          Cf[(size_t)row * N + col] = vv;
        } else if constexpr (OUT_MODE == 1) {
          Cb[(size_t)row * N + col] = (bf16_t)vv;
        } else if constexpr (OUT_MODE == 2) {
          int bb = row >> 11, s = row & (S_LEN - 1);
          int hh = col >> 6,  dd = col & (DHEAD - 1);
          Cb[(((size_t)(bb * NHEAD + hh)) * S_LEN + s) * DHEAD + dd] = (bf16_t)vv;
        } else {  // 3: V transposed [B,NH,DH,S]
          int bb = row >> 11, s = row & (S_LEN - 1);
          int hh = col >> 6,  dd = col & (DHEAD - 1);
          Cb[(((size_t)(bb * NHEAD + hh)) * DHEAD + dd) * S_LEN + s] = (bf16_t)vv;
        }
      }
    }
  }
}

// ---------------------------------------------------------------------------
// Flash attention: Q/K bf16 [B,NH,S,DH], V bf16 [B,NH,DH,S] (pre-transposed).
// Block: 128 threads = 4 wave32; 64 query rows/block (16 per wave).
// Key tiles of 32, double-buffered async LDS staging. ctx bf16 out in [B,S,H].
// ---------------------------------------------------------------------------
__global__ __launch_bounds__(128) void attn_flash_wmma(
    const bf16_t* __restrict__ qh, const bf16_t* __restrict__ kh,
    const bf16_t* __restrict__ vt, bf16_t* __restrict__ ctx) {
  __shared__ bf16_t Ks[2][32][72];     // [key][d]   (+8 pad)
  __shared__ bf16_t Vs[2][64][40];     // [d][key]   (+8 pad)
  __shared__ bf16_t Ps[4][16][32];     // per-wave P patch (C-layout -> A-layout)

  const int tid    = threadIdx.x;
  const int lane   = tid & 31;
  const int w      = tid >> 5;
  const int laneM  = lane & 15;
  const int laneHi = lane >> 4;
  const int bh     = blockIdx.y;
  const int bb     = bh >> 4;
  const int hh     = bh & (NHEAD - 1);
  const int q0     = blockIdx.x * 64;
  const float SCALE = 0.125f;          // 1/sqrt(64)

  // staging assignment (4 async b128 per thread per tile)
  const int kr = tid >> 2, kc = (tid & 3) * 16;  // K: key row 0..31
  const int vr = tid >> 1, vc = (tid & 1) * 16;  // V: d row 0..63, key chunk
  const bf16_t* Kbase = &kh[((size_t)bh * S_LEN + kr) * DHEAD + kc];
  const bf16_t* Vbase = &vt[((size_t)bh * DHEAD + vr) * S_LEN + vc];

  // ---- load Q fragments once (rows qw..qw+15, d=0..63) ----
  const int srow = q0 + w * 16 + laneM;
  const bf16_t* qp = &qh[((size_t)bh * S_LEN + srow) * DHEAD];
  FragBF aQ[2];
#pragma unroll
  for (int kd = 0; kd < 2; ++kd) {
    aQ[kd].f[0] = *(const float4*)(qp + kd * 32 + laneHi * 8);
    aQ[kd].f[1] = *(const float4*)(qp + kd * 32 + 16 + laneHi * 8);
  }

  v8f o[4] = {};
  float m[8], l[8];
#pragma unroll
  for (int r = 0; r < 8; ++r) { m[r] = -INFINITY; l[r] = 0.0f; }

  // preload key-tile 0
  async_b128(&Ks[0][kr][kc],     Kbase);
  async_b128(&Ks[0][kr][kc + 8], Kbase + 8);
  async_b128(&Vs[0][vr][vc],     Vbase);
  async_b128(&Vs[0][vr][vc + 8], Vbase + 8);

  const int nIter = S_LEN / 32;
  for (int it = 0; it < nIter; ++it) {
    const int cur = it & 1;
    wait_async0();
    __syncthreads();
    if (it + 1 < nIter) {
      const int kt1 = (it + 1) * 32;
      async_b128(&Ks[cur ^ 1][kr][kc],     Kbase + (size_t)kt1 * DHEAD);
      async_b128(&Ks[cur ^ 1][kr][kc + 8], Kbase + (size_t)kt1 * DHEAD + 8);
      async_b128(&Vs[cur ^ 1][vr][vc],     Vbase + kt1);
      async_b128(&Vs[cur ^ 1][vr][vc + 8], Vbase + kt1 + 8);
    }

    // ---- scores S = Q @ K^T (2 key groups x 2 d-steps) ----
    v8f sc[2] = {};
#pragma unroll
    for (int ni = 0; ni < 2; ++ni) {
      int keyr = ni * 16 + laneM;
#pragma unroll
      for (int kd = 0; kd < 2; ++kd) {
        FragBF bK;
        bK.f[0] = *(const float4*)&Ks[cur][keyr][kd * 32 + laneHi * 8];
        bK.f[1] = *(const float4*)&Ks[cur][keyr][kd * 32 + 16 + laneHi * 8];
        sc[ni] = wmma_bf16(aQ[kd].v, bK.v, sc[ni]);
      }
    }

    // ---- online softmax (rows = laneHi*8 + r; reduce across 16-lane half) ----
#pragma unroll
    for (int r = 0; r < 8; ++r) {
      float s0 = sc[0][r] * SCALE;
      float s1 = sc[1][r] * SCALE;
      float mx = fmaxf(s0, s1);
#pragma unroll
      for (int off = 8; off >= 1; off >>= 1)
        mx = fmaxf(mx, __shfl_xor(mx, off, 32));
      float mnew = fmaxf(m[r], mx);
      float corr = __expf(m[r] - mnew);
      float p0 = __expf(s0 - mnew);
      float p1 = __expf(s1 - mnew);
      float ps = p0 + p1;
#pragma unroll
      for (int off = 8; off >= 1; off >>= 1)
        ps += __shfl_xor(ps, off, 32);
      l[r] = l[r] * corr + ps;
      m[r] = mnew;
#pragma unroll
      for (int ni = 0; ni < 4; ++ni) o[ni][r] *= corr;
      int prow = laneHi * 8 + r;
      Ps[w][prow][laneM]      = (bf16_t)p0;
      Ps[w][prow][16 + laneM] = (bf16_t)p1;
    }

    // ---- O += P @ V ----
    FragBF aP;
    aP.f[0] = *(const float4*)&Ps[w][laneM][laneHi * 8];
    aP.f[1] = *(const float4*)&Ps[w][laneM][16 + laneHi * 8];
#pragma unroll
    for (int ni = 0; ni < 4; ++ni) {
      FragBF bV;
      int dcol = ni * 16 + laneM;
      bV.f[0] = *(const float4*)&Vs[cur][dcol][laneHi * 8];
      bV.f[1] = *(const float4*)&Vs[cur][dcol][16 + laneHi * 8];
      o[ni] = wmma_bf16(aP.v, bV.v, o[ni]);
    }
  }

  // ---- normalize and store ctx in [B,S,H] bf16 ----
#pragma unroll
  for (int r = 0; r < 8; ++r) {
    float inv = 1.0f / l[r];
    int s = q0 + w * 16 + laneHi * 8 + r;
    size_t base = ((size_t)bb * S_LEN + s) * HID + hh * DHEAD;
#pragma unroll
    for (int ni = 0; ni < 4; ++ni)
      ctx[base + ni * 16 + laneM] = (bf16_t)(o[ni][r] * inv);
  }
}

// ---------------------------------------------------------------------------
// Fused residual + LayerNorm (ddof=1 variance, eps added to std — matches ref)
// ---------------------------------------------------------------------------
__global__ __launch_bounds__(256) void ln_residual(
    const float* __restrict__ A, const float* __restrict__ Bv,
    const float* __restrict__ gamma, const float* __restrict__ beta,
    float* __restrict__ outF, bf16_t* __restrict__ outB) {
  const int row = blockIdx.x;
  const float* a = A + (size_t)row * HID;
  const float* b = Bv + (size_t)row * HID;
  float v[4];
  float s = 0.f, ss = 0.f;
#pragma unroll
  for (int i = 0; i < 4; ++i) {
    int idx = threadIdx.x + i * 256;
    v[i] = a[idx] + b[idx];
    s += v[i];
    ss += v[i] * v[i];
  }
#pragma unroll
  for (int off = 16; off >= 1; off >>= 1) {
    s  += __shfl_xor(s,  off, 32);
    ss += __shfl_xor(ss, off, 32);
  }
  __shared__ float red_s[8], red_ss[8];
  int wv = threadIdx.x >> 5, ln = threadIdx.x & 31;
  if (ln == 0) { red_s[wv] = s; red_ss[wv] = ss; }
  __syncthreads();
  s = 0.f; ss = 0.f;
#pragma unroll
  for (int i = 0; i < 8; ++i) { s += red_s[i]; ss += red_ss[i]; }
  float mean = s * (1.0f / HID);
  float var  = (ss - (float)HID * mean * mean) * (1.0f / (HID - 1));
  float inv  = 1.0f / (sqrtf(fmaxf(var, 0.0f)) + 1e-12f);
#pragma unroll
  for (int i = 0; i < 4; ++i) {
    int idx = threadIdx.x + i * 256;
    float y = gamma[idx] * (v[i] - mean) * inv + beta[idx];
    outF[(size_t)row * HID + idx] = y;
    if (outB) outB[(size_t)row * HID + idx] = (bf16_t)y;
  }
}

// ---------------------------------------------------------------------------
// Launch: full encoder layer
// ---------------------------------------------------------------------------
extern "C" void kernel_launch(void* const* d_in, const int* in_sizes, int n_in,
                              void* d_out, int out_size, void* d_ws, size_t ws_size,
                              hipStream_t stream) {
  const float* x     = (const float*)d_in[0];
  const float* Wq    = (const float*)d_in[1];
  const float* bq    = (const float*)d_in[2];
  const float* Wk    = (const float*)d_in[3];
  const float* bk    = (const float*)d_in[4];
  const float* Wv    = (const float*)d_in[5];
  const float* bv    = (const float*)d_in[6];
  const float* Wo    = (const float*)d_in[7];
  const float* bo    = (const float*)d_in[8];
  const float* ln1_g = (const float*)d_in[9];
  const float* ln1_b = (const float*)d_in[10];
  const float* W1    = (const float*)d_in[11];
  const float* b1    = (const float*)d_in[12];
  const float* W2    = (const float*)d_in[13];
  const float* b2    = (const float*)d_in[14];
  const float* ln2_g = (const float*)d_in[15];
  const float* ln2_b = (const float*)d_in[16];

  constexpr size_t E_H  = (size_t)NTOK * HID;    //  4,194,304
  constexpr size_t E_W  = (size_t)HID * HID;     //  1,048,576
  constexpr size_t E_W1 = (size_t)HID * FFDIM;   //  4,194,304
  constexpr size_t E_FF = (size_t)NTOK * FFDIM;  // 16,777,216

  // ---- workspace carve-up (~143 MB) ----
  bf16_t* xb   = (bf16_t*)d_ws;
  bf16_t* wqT  = xb + E_H;     // all weights stored transposed [N,K] bf16
  bf16_t* wkT  = wqT + E_W;
  bf16_t* wvT  = wkT + E_W;
  bf16_t* woT  = wvT + E_W;
  bf16_t* w1T  = woT + E_W;
  bf16_t* w2T  = w1T + E_W1;
  bf16_t* qhp  = w2T + E_W1;   // [B,NH,S,DH]
  bf16_t* khp  = qhp + E_H;    // [B,NH,S,DH]
  bf16_t* vhp  = khp + E_H;    // [B,NH,DH,S]  (transposed for attention)
  bf16_t* ctxb = vhp + E_H;    // [B,S,H]
  bf16_t* hb   = ctxb + E_H;   // LN1 output, bf16
  bf16_t* f1b  = hb + E_H;     // FFN intermediate [NTOK,FF]
  float*  attnf = (float*)(f1b + E_FF);
  float*  hf    = attnf + E_H;
  float*  f2f   = attnf;       // reuse: attnf dead after LN1

  // ---- convert / transpose ----
  cvt_f32_bf16<<<2048, 256, 0, stream>>>(x, xb, (int)E_H);
  dim3 gW(HID / 32, HID / 32);
  cvt_transpose_bf16<<<gW, 256, 0, stream>>>(Wq, wqT, HID, HID);
  cvt_transpose_bf16<<<gW, 256, 0, stream>>>(Wk, wkT, HID, HID);
  cvt_transpose_bf16<<<gW, 256, 0, stream>>>(Wv, wvT, HID, HID);
  cvt_transpose_bf16<<<gW, 256, 0, stream>>>(Wo, woT, HID, HID);
  cvt_transpose_bf16<<<dim3(FFDIM / 32, HID / 32), 256, 0, stream>>>(W1, w1T, HID, FFDIM);
  cvt_transpose_bf16<<<dim3(HID / 32, FFDIM / 32), 256, 0, stream>>>(W2, w2T, FFDIM, HID);

  // ---- Q/K/V projections ----
  dim3 gproj(HID / 64, NTOK / 128);  // (16,32)
  gemm_bf16_wmma<2, false><<<gproj, 256, 0, stream>>>(xb, wqT, bq, nullptr, qhp, NTOK, HID, HID);
  gemm_bf16_wmma<2, false><<<gproj, 256, 0, stream>>>(xb, wkT, bk, nullptr, khp, NTOK, HID, HID);
  gemm_bf16_wmma<3, false><<<gproj, 256, 0, stream>>>(xb, wvT, bv, nullptr, vhp, NTOK, HID, HID);

  // ---- attention ----
  attn_flash_wmma<<<dim3(S_LEN / 64, 2 * NHEAD), 128, 0, stream>>>(qhp, khp, vhp, ctxb);

  // ---- output projection ----
  gemm_bf16_wmma<0, false><<<gproj, 256, 0, stream>>>(ctxb, woT, bo, attnf, nullptr, NTOK, HID, HID);

  // ---- residual + LN1 (f32 + bf16 copy) ----
  ln_residual<<<NTOK, 256, 0, stream>>>(x, attnf, ln1_g, ln1_b, hf, hb);

  // ---- FFN ----
  dim3 gff1(FFDIM / 64, NTOK / 128);  // (64,32)
  gemm_bf16_wmma<1, true><<<gff1, 256, 0, stream>>>(hb, w1T, b1, nullptr, f1b, NTOK, FFDIM, HID);
  gemm_bf16_wmma<0, false><<<gproj, 256, 0, stream>>>(f1b, w2T, b2, f2f, nullptr, NTOK, HID, FFDIM);

  // ---- residual + LN2 -> output ----
  ln_residual<<<NTOK, 256, 0, stream>>>(hf, f2f, ln2_g, ln2_b, (float*)d_out, nullptr);
}
